// FlashLlamaAttention_17119739642654
// MI455X (gfx1250) — compile-verified, matrix operand order checked by reading, not run on hardware
//
#include <hip/hip_runtime.h>

// ---------------- problem constants (from reference) ----------------
constexpr int Hh   = 32;
constexpr int Dd   = 128;
constexpr int HIDc = 4096;          // H*D
constexpr int Bb   = 4;
constexpr int Ss   = 1024;
constexpr int Tt   = Bb * Ss;       // 4096
constexpr int N3   = 3 * HIDc;      // 12288
constexpr float SCALE_ATTN = 0.08838834764831845f; // D^-0.5

// ---------------- types ----------------
typedef __bf16 bf16;
typedef __attribute__((ext_vector_type(16))) __bf16 v16bf;
typedef __attribute__((ext_vector_type(8)))  float  v8f;
typedef __attribute__((ext_vector_type(4)))  unsigned int u32x4;

union Frag16 { u32x4 q[2]; v16bf v; };      // 32 bytes, POD
union Pack8  { u32x4 q;    bf16  e[8]; };   // 16 bytes, POD

// ---------------- Tensor Data Mover (gfx1250) ----------------
#if __has_builtin(__builtin_amdgcn_tensor_load_to_lds) && \
    __has_builtin(__builtin_amdgcn_s_wait_tensorcnt)
#define HAVE_TDM 1
typedef __attribute__((ext_vector_type(4))) unsigned int tdm_u4;
typedef __attribute__((ext_vector_type(8))) int          tdm_i8;
typedef __attribute__((ext_vector_type(4))) int          tdm_i4;

// Issue a 2-D bf16 tile DMA: global (tile_d1 rows x tile_d0 elems, row stride
// `stride_elems`) -> LDS at byte offset lds_off, with optional per-row LDS
// padding (pad_interval/pad_amount codes per D# group1 spec) so the LDS image
// lands with a padded pitch. Descriptor "tensor" == the tile window (all
// tiles here are fully in-bounds). This toolchain's builtin takes 6 args:
// (u32x4 g0, i32x8 g1, i32x4 g2, i32x4 g3, i32x8 extra, i32 cpol).
__device__ __forceinline__ void tdm_load_2d_bf16(
    const void* gsrc, unsigned int lds_off,
    unsigned int tile_d0, unsigned int tile_d1,
    unsigned long long stride_elems,
    unsigned int pad_interval_code, unsigned int pad_amount_code, bool pad_en)
{
    const unsigned long long ga = (unsigned long long)gsrc;
    tdm_u4 g0;
    g0[0] = 1u;                                            // count=1 (user D#)
    g0[1] = lds_off;                                       // lds_addr (bytes)
    g0[2] = (unsigned int)(ga & 0xffffffffull);            // global_addr[31:0]
    g0[3] = (unsigned int)((ga >> 32) & 0x1ffffffull)      // global_addr[56:32]
          | (2u << 30);                                    // type = 2 ("image")
    tdm_i8 g1;
    g1[0] = (int)((1u << 16)                               // data_size=1 -> 2B
          | (pad_en ? (1u << 20) : 0u)                     // pad_enable
          | ((pad_interval_code & 7u) << 22)
          | ((pad_amount_code & 127u) << 25));
    g1[1] = (int)((tile_d0 & 0xffffu) << 16);              // tensor_dim0[15:0]
    g1[2] = (int)(((tile_d0 >> 16) & 0xffffu)              // tensor_dim0[31:16]
          | ((tile_d1 & 0xffffu) << 16));                  // tensor_dim1[15:0]
    g1[3] = (int)(((tile_d1 >> 16) & 0xffffu)              // tensor_dim1[31:16]
          | ((tile_d0 & 0xffffu) << 16));                  // tile_dim0
    g1[4] = (int)(tile_d1 & 0xffffu);                      // tile_dim1 (tile_dim2=0)
    g1[5] = (int)(stride_elems & 0xffffffffull);           // dim0_stride[31:0]
    g1[6] = (int)((stride_elems >> 32) & 0xffffull);       // dim0_stride[47:32]
    g1[7] = 0;
    const tdm_i4 gz4 = {0, 0, 0, 0};                       // 2-D: groups 2/3 zero
    const tdm_i8 gz8 = {0, 0, 0, 0, 0, 0, 0, 0};
    __builtin_amdgcn_tensor_load_to_lds(g0, g1, gz4, gz4, gz8, 0);
}
__device__ __forceinline__ unsigned int lds_offset_of(const void* p) {
    // LDS aperture lives entirely in addr[63:32]; addr[31:0] is the LDS byte address.
    return (unsigned int)(unsigned long long)p;
}
#endif

// Load a 16x32 bf16 fragment (A-layout / transposed-B) from an LDS tile stored
// row-major with `pitch` bf16 elements per row (pitch % 8 == 0 for 16B loads).
// CDNA5 16-bit layout: lanes 0-15 -> K {0..7,16..23}, lanes 16-31 -> K {8..15,24..31}.
__device__ __forceinline__ v16bf load_frag(const bf16* base, int pitch,
                                           int rowbase, int kbase, int lane) {
    const int r  = rowbase + (lane & 15);
    const int k0 = kbase + ((lane >> 4) << 3);
    const bf16* p = base + r * pitch + k0;
    Frag16 f;
    f.q[0] = *(const u32x4*)(p);
    f.q[1] = *(const u32x4*)(p + 16);
    return f.v;
}

__device__ __forceinline__ v8f wmma_bf16(v16bf a, v16bf b, v8f c) {
    return __builtin_amdgcn_wmma_f32_16x16x32_bf16(
        false, a, false, b, (short)0, c, false, false);
}

// ---------------- f32 -> bf16 convert ----------------
__global__ void cvt_f32_bf16(const float* __restrict__ src,
                             bf16* __restrict__ dst, size_t n) {
    size_t i = (size_t)blockIdx.x * blockDim.x + threadIdx.x;
    const size_t stride = (size_t)gridDim.x * blockDim.x;
    for (; i < n; i += stride) dst[i] = (bf16)src[i];
}

// ---------------- bf16 WMMA GEMM: C(f32) = A(bf16,MxK) * B(bf16,KxN) ----------------
// Block tile 128x128, K-step 64, 8 waves: wave (wm 0..3, wn 0..1) owns 32x64.
__global__ __launch_bounds__(256)
void gemm_bf16_wmma(const bf16* __restrict__ A, const bf16* __restrict__ Bm,
                    float* __restrict__ C, int M, int N, int K) {
    __shared__ __align__(16) bf16 As[128][72];    // row-major [m][k], pitch 72
    __shared__ __align__(16) bf16 Bst[128][72];   // transposed [n][k], pitch 72

    const int tid  = threadIdx.x;
    const int lane = tid & 31;
    const int wid  = tid >> 5;
    const int wm   = wid & 3;
    const int wn   = wid >> 2;
    const int mb   = blockIdx.y * 128;
    const int nb   = blockIdx.x * 128;

    v8f acc[2][4];
    const v8f vzero = {0.f,0.f,0.f,0.f,0.f,0.f,0.f,0.f};
#pragma unroll
    for (int i = 0; i < 2; ++i)
#pragma unroll
        for (int j = 0; j < 4; ++j) acc[i][j] = vzero;

    for (int kb = 0; kb < K; kb += 64) {
        // ---- stage A tile 128x64 via TDM (row 64 elems=128B=32 DW -> interval
        //      code 4; pad 8 elems=4 DW -> amount code 3 => pitch 72) ----
#if defined(HAVE_TDM)
        if (wid == 0) {
            tdm_load_2d_bf16(A + (size_t)mb * K + kb, lds_offset_of(&As[0][0]),
                             64u, 128u, (unsigned long long)K, 4u, 3u, true);
        }
#else
#pragma unroll
        for (int j = 0; j < 4; ++j) {
            const int idx = tid * 4 + j;          // 0..1023 uint4s
            const int row = idx >> 3;             // 8 uint4 per 64-elem row
            const int c8  = (idx & 7) << 3;
            const u32x4 d = *(const u32x4*)(A + (size_t)(mb + row) * K + kb + c8);
            *(u32x4*)(&As[row][c8]) = d;
        }
#endif
        // ---- stage B tile 64x128, transposed into Bst[n][k] (manual) ----
#pragma unroll
        for (int j = 0; j < 4; ++j) {
            const int idx = tid * 4 + j;          // 0..1023 uint4s
            const int kr  = idx >> 4;             // 16 uint4 per 128-elem row
            const int c8  = (idx & 15) << 3;
            Pack8 d;
            d.q = *(const u32x4*)(Bm + (size_t)(kb + kr) * N + nb + c8);
#pragma unroll
            for (int t = 0; t < 8; ++t) Bst[c8 + t][kr] = d.e[t];
        }
        if (kb + 64 < K) {
            __builtin_prefetch(A + (size_t)(mb + (tid >> 1)) * K + kb + 64, 0, 1);
            __builtin_prefetch(Bm + (size_t)(kb + 64 + (tid >> 2)) * N + nb, 0, 1);
        }
#if defined(HAVE_TDM)
        if (wid == 0) __builtin_amdgcn_s_wait_tensorcnt(0);
#endif
        __syncthreads();

#pragma unroll
        for (int ks = 0; ks < 2; ++ks) {
            v16bf afr[2], bfr[4];
#pragma unroll
            for (int ti = 0; ti < 2; ++ti)
                afr[ti] = load_frag(&As[0][0], 72, wm * 32 + ti * 16, ks * 32, lane);
#pragma unroll
            for (int tj = 0; tj < 4; ++tj)
                bfr[tj] = load_frag(&Bst[0][0], 72, wn * 64 + tj * 16, ks * 32, lane);
#pragma unroll
            for (int ti = 0; ti < 2; ++ti)
#pragma unroll
                for (int tj = 0; tj < 4; ++tj)
                    acc[ti][tj] = wmma_bf16(afr[ti], bfr[tj], acc[ti][tj]);
        }
        __syncthreads();
    }

    const int rbase = ((lane >> 4) << 3);
    const int cbase = (lane & 15);
#pragma unroll
    for (int ti = 0; ti < 2; ++ti)
#pragma unroll
        for (int tj = 0; tj < 4; ++tj)
#pragma unroll
            for (int i = 0; i < 8; ++i) {
                const int row = mb + wm * 32 + ti * 16 + rbase + i;
                const int col = nb + wn * 64 + tj * 16 + cbase;
                C[(size_t)row * N + col] = acc[ti][tj][i];
            }
}

// ---------------- RoPE + cache scatter + bf16 Q/K/V staging ----------------
__global__ void rope_cache_kernel(const float* __restrict__ qkv,
                                  const float* __restrict__ cosT,
                                  const float* __restrict__ sinT,
                                  const int* __restrict__ slots,
                                  bf16* __restrict__ qb, bf16* __restrict__ kb,
                                  bf16* __restrict__ vb,
                                  float* __restrict__ kcache,
                                  float* __restrict__ vcache) {
    const int idx = blockIdx.x * blockDim.x + threadIdx.x;   // over T*H*64
    const int dp  = idx & 63;
    const int h   = (idx >> 6) & (Hh - 1);
    const int t   = idx >> 11;
    if (t >= Tt) return;

    const float c = cosT[t * 64 + dp];
    const float s = sinT[t * 64 + dp];

    const size_t base = (size_t)t * N3 + h * Dd + dp * 2;
    const float q1 = qkv[base],            q2 = qkv[base + 1];
    const float k1 = qkv[base + HIDc],     k2 = qkv[base + HIDc + 1];
    const float v1 = qkv[base + 2 * HIDc], v2 = qkv[base + 2 * HIDc + 1];

    const float qr1 = q1 * c - q2 * s, qr2 = q2 * c + q1 * s;
    const float kr1 = k1 * c - k2 * s, kr2 = k2 * c + k1 * s;

    const size_t o = (size_t)t * HIDc + h * Dd + dp * 2;
    qb[o] = (bf16)qr1; qb[o + 1] = (bf16)qr2;
    kb[o] = (bf16)kr1; kb[o + 1] = (bf16)kr2;
    vb[o] = (bf16)v1;  vb[o + 1] = (bf16)v2;

    const int slot = slots[t];
    const size_t co = (size_t)slot * HIDc + h * Dd + dp * 2;
    kcache[co] = kr1; kcache[co + 1] = kr2;
    vcache[co] = v1;  vcache[co + 1] = v2;
}

// ---------------- causal flash attention (bf16 WMMA, f32 online softmax) ----------------
// grid: (S/128, H, B); block 256 = 8 waves; wave w owns q rows [w*16, w*16+16).
__global__ __launch_bounds__(256)
void flash_attn_kernel(const bf16* __restrict__ Q, const bf16* __restrict__ Kv,
                       const bf16* __restrict__ Vv, bf16* __restrict__ Ctx) {
    __shared__ __align__(16) bf16 Qs[128][136];    // [q_local][d]
    __shared__ __align__(16) bf16 Ks[32][136];     // [key_local][d]
    __shared__ __align__(16) bf16 Vst[128][40];    // [d][key_local]
    __shared__ __align__(16) bf16 Ps[8][16][40];   // per-wave P tile [q][key]

    const int qb = blockIdx.x, h = blockIdx.y, b = blockIdx.z;
    const int tid = threadIdx.x, lane = tid & 31, wid = tid >> 5;

    // stage Q block: 128 rows x 128 d (row 128 elems=256B=64 DW -> interval
    // code 5; pad 4 DW -> amount code 3 => pitch 136)
#if defined(HAVE_TDM)
    if (wid == 0) {
        tdm_load_2d_bf16(Q + ((size_t)(b * Ss + qb * 128)) * HIDc + h * Dd,
                         lds_offset_of(&Qs[0][0]),
                         128u, 128u, (unsigned long long)HIDc, 5u, 3u, true);
    }
#else
#pragma unroll
    for (int j = 0; j < 8; ++j) {
        const int idx = tid + j * 256;
        const int r   = idx >> 4;
        const int c8  = (idx & 15) << 3;
        const bf16* src = Q + ((size_t)(b * Ss + qb * 128 + r)) * HIDc + h * Dd + c8;
        *(u32x4*)(&Qs[r][c8]) = *(const u32x4*)src;
    }
#endif

    float m_i[8], l_i[8];
    v8f acc[8];
    const v8f vzero = {0.f,0.f,0.f,0.f,0.f,0.f,0.f,0.f};
#pragma unroll
    for (int i = 0; i < 8; ++i) { m_i[i] = -3.0e38f; l_i[i] = 0.f; acc[i] = vzero; }

    const int qrow0 = qb * 128 + wid * 16 + ((lane >> 4) << 3);
    const int nkb   = (qb + 1) * 4;   // causal bound, key blocks of 32

    for (int kblk = 0; kblk < nkb; ++kblk) {
        const int kbase = kblk * 32;
        __syncthreads();   // prior iteration's WMMAs done before K/V restage
        // stage K tile 32x128 (row-major)
#if defined(HAVE_TDM)
        if (wid == 0) {
            tdm_load_2d_bf16(Kv + ((size_t)(b * Ss + kbase)) * HIDc + h * Dd,
                             lds_offset_of(&Ks[0][0]),
                             128u, 32u, (unsigned long long)HIDc, 5u, 3u, true);
        }
#else
#pragma unroll
        for (int j = 0; j < 2; ++j) {
            const int idx = tid * 2 + j;
            const int r = idx >> 4, c8 = (idx & 15) << 3;
            const bf16* src = Kv + ((size_t)(b * Ss + kbase + r)) * HIDc + h * Dd + c8;
            *(u32x4*)(&Ks[r][c8]) = *(const u32x4*)src;
        }
#endif
        // stage V tile 32x128 transposed -> Vst[d][key] (manual)
#pragma unroll
        for (int j = 0; j < 2; ++j) {
            const int idx = tid * 2 + j;
            const int r = idx >> 4, c8 = (idx & 15) << 3;
            Pack8 d;
            d.q = *(const u32x4*)(Vv + ((size_t)(b * Ss + kbase + r)) * HIDc + h * Dd + c8);
#pragma unroll
            for (int t = 0; t < 8; ++t) Vst[c8 + t][r] = d.e[t];
        }
#if defined(HAVE_TDM)
        if (wid == 0) __builtin_amdgcn_s_wait_tensorcnt(0);
#endif
        __syncthreads();

        // S tiles: 16 q rows x 32 keys, contraction over D=128
        v8f sfr[2] = {vzero, vzero};
#pragma unroll
        for (int kt = 0; kt < 2; ++kt)
#pragma unroll
            for (int ks = 0; ks < 4; ++ks) {
                v16bf a  = load_frag(&Qs[0][0], 136, wid * 16, ks * 32, lane);
                v16bf bb = load_frag(&Ks[0][0], 136, kt * 16,  ks * 32, lane);
                sfr[kt] = wmma_bf16(a, bb, sfr[kt]);
            }

        const int k0g = kbase + (lane & 15);
        float alpha[8];
#pragma unroll
        for (int i = 0; i < 8; ++i) {
            const int qg = qrow0 + i;
            float s0 = (k0g      <= qg) ? sfr[0][i] * SCALE_ATTN : -3.0e38f;
            float s1 = (k0g + 16 <= qg) ? sfr[1][i] * SCALE_ATTN : -3.0e38f;
            float t = fmaxf(s0, s1);
            t = fmaxf(t, __shfl_xor(t, 1));
            t = fmaxf(t, __shfl_xor(t, 2));
            t = fmaxf(t, __shfl_xor(t, 4));
            t = fmaxf(t, __shfl_xor(t, 8));
            const float mnew = fmaxf(m_i[i], t);
            alpha[i] = __expf(m_i[i] - mnew);
            m_i[i] = mnew;
            const float p0 = __expf(s0 - mnew);
            const float p1 = __expf(s1 - mnew);
            float rs = p0 + p1;
            rs += __shfl_xor(rs, 1);
            rs += __shfl_xor(rs, 2);
            rs += __shfl_xor(rs, 4);
            rs += __shfl_xor(rs, 8);
            l_i[i] = l_i[i] * alpha[i] + rs;
            sfr[0][i] = p0; sfr[1][i] = p1;
        }
#pragma unroll
        for (int nt = 0; nt < 8; ++nt)
#pragma unroll
            for (int i = 0; i < 8; ++i) acc[nt][i] *= alpha[i];

        // stash P as bf16 (wave-private LDS; same-wave DS ops stay ordered)
        const int prow = ((lane >> 4) << 3);
#pragma unroll
        for (int kt = 0; kt < 2; ++kt)
#pragma unroll
            for (int i = 0; i < 8; ++i)
                Ps[wid][prow + i][kt * 16 + (lane & 15)] = (bf16)sfr[kt][i];

        // O += P * V
        v16bf pa = load_frag(&Ps[wid][0][0], 40, 0, 0, lane);
#pragma unroll
        for (int nt = 0; nt < 8; ++nt) {
            v16bf vb2 = load_frag(&Vst[0][0], 40, nt * 16, 0, lane);
            acc[nt] = wmma_bf16(pa, vb2, acc[nt]);
        }
    }

    float inv[8];
#pragma unroll
    for (int i = 0; i < 8; ++i) inv[i] = 1.0f / l_i[i];
#pragma unroll
    for (int nt = 0; nt < 8; ++nt)
#pragma unroll
        for (int i = 0; i < 8; ++i) {
            const int qg = qrow0 + i;
            const int d  = nt * 16 + (lane & 15);
            Ctx[((size_t)(b * Ss + qg)) * HIDc + h * Dd + d] = (bf16)(acc[nt][i] * inv[i]);
        }
}

// ---------------- launcher ----------------
extern "C" void kernel_launch(void* const* d_in, const int* in_sizes, int n_in,
                              void* d_out, int out_size, void* d_ws, size_t ws_size,
                              hipStream_t stream) {
    (void)in_sizes; (void)n_in; (void)out_size; (void)ws_size;
    const float* hidden = (const float*)d_in[0];
    const float* cosT   = (const float*)d_in[1];
    const float* sinT   = (const float*)d_in[2];
    const float* wqkv   = (const float*)d_in[3];
    const float* wo     = (const float*)d_in[4];
    const int*   slots  = (const int*)d_in[7];

    float* outp   = (float*)d_out;
    float* kcache = outp + (size_t)Tt * HIDc;
    float* vcache = kcache + (size_t)Tt * HIDc;

    char* w = (char*)d_ws;
    size_t off = 0;
    bf16*  hid_bf  = (bf16*)(w + off);  off += (size_t)Tt * HIDc * 2;
    bf16*  wqkv_bf = (bf16*)(w + off);  off += (size_t)HIDc * N3 * 2;
    bf16*  wo_bf   = (bf16*)(w + off);  off += (size_t)HIDc * HIDc * 2;
    float* qkv_f   = (float*)(w + off); off += (size_t)Tt * N3 * 4;
    bf16*  q_bf    = (bf16*)(w + off);  off += (size_t)Tt * HIDc * 2;
    bf16*  k_bf    = (bf16*)(w + off);  off += (size_t)Tt * HIDc * 2;
    bf16*  v_bf    = (bf16*)(w + off);  off += (size_t)Tt * HIDc * 2;
    bf16*  ctx_bf  = (bf16*)(w + off);  off += (size_t)Tt * HIDc * 2;

    cvt_f32_bf16<<<2048, 256, 0, stream>>>(hidden, hid_bf, (size_t)Tt * HIDc);
    cvt_f32_bf16<<<4096, 256, 0, stream>>>(wqkv, wqkv_bf, (size_t)HIDc * N3);
    cvt_f32_bf16<<<2048, 256, 0, stream>>>(wo, wo_bf, (size_t)HIDc * HIDc);

    gemm_bf16_wmma<<<dim3(N3 / 128, Tt / 128), 256, 0, stream>>>(
        hid_bf, wqkv_bf, qkv_f, Tt, N3, HIDc);

    rope_cache_kernel<<<(Tt * Hh * 64) / 256, 256, 0, stream>>>(
        qkv_f, cosT, sinT, slots, q_bf, k_bf, v_bf, kcache, vcache);

    flash_attn_kernel<<<dim3(Ss / 128, Hh, Bb), 256, 0, stream>>>(
        q_bf, k_bf, v_bf, ctx_bf);

    gemm_bf16_wmma<<<dim3(HIDc / 128, Tt / 128), 256, 0, stream>>>(
        ctx_bf, wo_bf, outp, Tt, HIDc, HIDc);
}
